// SlotViTArc_88776974009069
// MI455X (gfx1250) — compile-verified, matrix operand order checked by reading, not run on hardware
//
#include <hip/hip_runtime.h>
#include <math.h>

// ---------------- problem constants ----------------
#define BB      4
#define CC      3
#define IMGSZ   256
#define PSZ     16
#define HPATCH  16
#define HW      256
#define DIM     512
#define DEPTH   6
#define NHEADS  8
#define DH      64
#define NSLOT   50
#define SHEADS  4
#define SDH     64
#define SI      (SHEADS*SDH)     // 256
#define ITERS   3
#define RPD     128
#define FFD     2048
#define NTOK    (NSLOT + HW)     // 306
#define PATCH_K (CC*PSZ*PSZ)     // 768
#define SIMLD   320              // padded row stride for bf16 attention probs (16B-aligned rows)

typedef __attribute__((ext_vector_type(16))) __bf16 bf16x16;
typedef __attribute__((ext_vector_type(8)))  float  floatx8;

// ---------------- scalar helpers ----------------
__device__ __forceinline__ __bf16 f2bf(float f) {
  union { float f; unsigned u; } v; v.f = f;
  unsigned r = (v.u + 0x7FFFu + ((v.u >> 16) & 1u)) >> 16;   // RNE
  union { unsigned short s; __bf16 b; } o; o.s = (unsigned short)r;
  return o.b;
}
__device__ __forceinline__ float bf2f(__bf16 b) {
  union { unsigned short s; __bf16 b; } i; i.b = b;
  union { unsigned u; float f; } o; o.u = ((unsigned)i.s) << 16;
  return o.f;
}
__device__ __forceinline__ float sigmoidf_(float x) { return 1.f / (1.f + __expf(-x)); }
__device__ __forceinline__ float apply_act(float x, int act) {
  if (act == 1) return x > 0.f ? x : 0.f;                                   // relu
  if (act == 2) {                                                           // gelu (tanh approx)
    float x3 = x * x * x;
    return 0.5f * x * (1.f + tanhf(0.7978845608028654f * (x + 0.044715f * x3)));
  }
  if (act == 3) return x * sigmoidf_(x);                                    // silu
  return x;
}

// Low 32 bits of a generic pointer to LDS == LDS byte offset (ISA 10.2 aperture mapping).
__device__ __forceinline__ unsigned lds_addr_of(const void* p) {
  return (unsigned)(size_t)p;
}
// CDNA5 async global->LDS copy, 16B per lane, tracked by ASYNCcnt.
__device__ __forceinline__ void async_b128(unsigned lds_byte, const __bf16* g) {
  asm volatile("global_load_async_to_lds_b128 %0, %1, off" :: "v"(lds_byte), "v"(g) : "memory");
}
__device__ __forceinline__ void wait_async0() {
  asm volatile("s_wait_asynccnt 0" ::: "memory");
}

// Fragment gather per CDNA5 ISA 7.12.2 (16-bit A/B):
//   lane L: row = L&15 ; K(e) = e + 8*(e>>3) + 8*(L>>4)
// For tiles stored [row][K] (A tiles, and B tiles when transB=1).
__device__ __forceinline__ bf16x16 load_frag(const __bf16* row0, int ldk, int hlf, int l16) {
  const __bf16* p = row0 + l16 * ldk + 8 * hlf;
  bf16x16 f;
#pragma unroll
  for (int e = 0; e < 8; e++)  f[e] = p[e];
#pragma unroll
  for (int e = 8; e < 16; e++) f[e] = p[e + 8];
  return f;
}
// Same K pattern but tile stored [K][N] (B tiles when transB=0): strided column gather.
__device__ __forceinline__ bf16x16 load_frag_kn(const __bf16* col0, int ldn, int hlf) {
  const __bf16* p = col0 + (8 * hlf) * ldn;
  bf16x16 f;
#pragma unroll
  for (int e = 0; e < 8; e++)  f[e] = p[e * ldn];
#pragma unroll
  for (int e = 8; e < 16; e++) f[e] = p[(e + 8) * ldn];
  return f;
}

// ---------------- generic strided-batched WMMA GEMM (bf16 in, f32 accum) ----------------
// C(f32, optional) / C16(bf16, optional) = act(alpha * A@B(^T) + bias_n + Res)
#define TMM 64
#define TNN 64
#define TKK 32
#define LDKP 40   // [row][K] padded stride: 80B rows, 16B-aligned quads
#define LDNP 72   // [K][N]  padded stride: 144B rows, 16B-aligned quads

__global__ __launch_bounds__(128)
void gemm_kernel(const __bf16* __restrict__ A, int lda, long sAb, long sAh,
                 const __bf16* __restrict__ B, int ldb, long sBb, long sBh,
                 const float* __restrict__ bias,
                 const float* __restrict__ Res, int ldr, long sRb, long sRh,
                 float* __restrict__ C, __bf16* __restrict__ C16,
                 int ldc, long sCb, long sCh,
                 int M, int N, int K, int batchH, float alpha, int act, int transB)
{
  __shared__ __bf16 As[TMM * LDKP];
  __shared__ __bf16 Bs[TMM * LDKP];    // [N][K]@LDKP (transB=1) or [K][N]@LDNP (transB=0)
  const int z  = blockIdx.z;
  const int zb = z / batchH, zh = z % batchH;
  A += (long)zb * sAb + (long)zh * sAh;
  B += (long)zb * sBb + (long)zh * sBh;
  if (C)   C   += (long)zb * sCb + (long)zh * sCh;
  if (C16) C16 += (long)zb * sCb + (long)zh * sCh;
  if (Res) Res += (long)zb * sRb + (long)zh * sRh;

  const int m0 = blockIdx.y * TMM;
  const int n0 = blockIdx.x * TNN;
  const int tid  = threadIdx.x;
  const int wid  = tid >> 5;
  const int lane = tid & 31;
  const int hlf  = lane >> 4;
  const int l16  = lane & 15;
  const int wm = (wid >> 1) * 32;
  const int wn = (wid & 1) * 32;
  const unsigned asBase = lds_addr_of(As);
  const unsigned bsBase = lds_addr_of(Bs);
  const __bf16 zb16 = f2bf(0.f);

  const bool fullA = (m0 + TMM <= M) && ((lda & 7) == 0);
  const bool fullB = (n0 + TNN <= N) && ((ldb & 7) == 0);

  const floatx8 zero = {0.f, 0.f, 0.f, 0.f, 0.f, 0.f, 0.f, 0.f};
  floatx8 acc[2][2] = {{zero, zero}, {zero, zero}};

  for (int k0 = 0; k0 < K; k0 += TKK) {
    const bool fullK = (k0 + TKK <= K);
    // ---- A tile: [TMM][TKK] rows of 64B = 4 x 16B chunks ----
    if (fullA && fullK) {
      const __bf16* ga = A + (long)m0 * lda + k0;
      for (int c = tid; c < TMM * 4; c += 128) {
        int r = c >> 2, q = c & 3;
        async_b128(asBase + r * (LDKP * 2) + q * 16, ga + (long)r * lda + q * 8);
      }
    } else {
      for (int t = tid; t < TMM * TKK; t += 128) {
        int r = t >> 5, c2 = t & 31;
        int gm = m0 + r, gk = k0 + c2;
        As[r * LDKP + c2] = (gm < M && gk < K) ? A[(long)gm * lda + gk] : zb16;
      }
    }
    // ---- B tile ----
    if (transB) {            // global [N][K] -> LDS [N][K]
      if (fullB && fullK) {
        const __bf16* gb = B + (long)n0 * ldb + k0;
        for (int c = tid; c < TNN * 4; c += 128) {
          int r = c >> 2, q = c & 3;
          async_b128(bsBase + r * (LDKP * 2) + q * 16, gb + (long)r * ldb + q * 8);
        }
      } else {
        for (int t = tid; t < TNN * TKK; t += 128) {
          int r = t >> 5, c2 = t & 31;
          int gn = n0 + r, gk = k0 + c2;
          Bs[r * LDKP + c2] = (gn < N && gk < K) ? B[(long)gn * ldb + gk] : zb16;
        }
      }
    } else {                 // global [K][N] -> LDS [K][N], rows of 128B = 8 x 16B chunks
      if (fullB && fullK) {
        const __bf16* gb = B + (long)k0 * ldb + n0;
        for (int c = tid; c < TKK * 8; c += 128) {
          int r = c >> 3, q = c & 7;
          async_b128(bsBase + r * (LDNP * 2) + q * 16, gb + (long)r * ldb + q * 8);
        }
      } else {
        for (int t = tid; t < TKK * TNN; t += 128) {
          int r = t >> 6, c2 = t & 63;
          int gk = k0 + r, gn = n0 + c2;
          Bs[r * LDNP + c2] = (gk < K && gn < N) ? B[(long)gk * ldb + gn] : zb16;
        }
      }
    }
    wait_async0();
    __syncthreads();

#pragma unroll
    for (int mi = 0; mi < 2; mi++) {
      bf16x16 af = load_frag(&As[(wm + mi * 16) * LDKP], LDKP, hlf, l16);
#pragma unroll
      for (int ni = 0; ni < 2; ni++) {
        bf16x16 bfr = transB ? load_frag(&Bs[(wn + ni * 16) * LDKP], LDKP, hlf, l16)
                             : load_frag_kn(&Bs[wn + ni * 16 + l16], LDNP, hlf);
        acc[mi][ni] = __builtin_amdgcn_wmma_f32_16x16x32_bf16(
            false, af, false, bfr, (short)0, acc[mi][ni], false, false);
      }
    }
    __syncthreads();
  }

  // epilogue: C layout lane L: N = L&15, VGPR v: M = v + 8*(L>>4)
#pragma unroll
  for (int mi = 0; mi < 2; mi++)
#pragma unroll
    for (int ni = 0; ni < 2; ni++) {
      int nc = n0 + wn + ni * 16 + l16;
      if (nc >= N) continue;
      float bv = bias ? bias[nc] : 0.f;
#pragma unroll
      for (int v = 0; v < 8; v++) {
        int mr = m0 + wm + mi * 16 + v + 8 * hlf;
        if (mr >= M) continue;
        float r = alpha * acc[mi][ni][v] + bv;
        if (Res) r += Res[(long)mr * ldr + nc];
        r = apply_act(r, act);
        if (C)   C[(long)mr * ldc + nc] = r;
        if (C16) C16[(long)mr * ldc + nc] = f2bf(r);
      }
    }
}

// ---------------- conversion kernels ----------------
__global__ void convert_kernel(const float* __restrict__ s, __bf16* __restrict__ d, long n) {
  long idx = (long)blockIdx.x * blockDim.x + threadIdx.x;
  if (idx < n) d[idx] = f2bf(s[idx]);
}
__global__ void rpw2t_kernel(const float* __restrict__ s, __bf16* __restrict__ d) {
  int idx = blockIdx.x * blockDim.x + threadIdx.x;
  if (idx >= RPD * RPD) return;
  int n = idx >> 7, k = idx & 127;
  d[idx] = f2bf(s[k * RPD + n]);        // d[n][k] = w2[k][n]
}

// ---------------- elementwise / norm / softmax kernels ----------------
__global__ void patchify_kernel(const float* __restrict__ img, __bf16* __restrict__ Xp) {
  long idx = (long)blockIdx.x * blockDim.x + threadIdx.x;
  if (idx >= (long)BB * HW * PATCH_K) return;
  long row = idx / PATCH_K; int kk = (int)(idx % PATCH_K);
  int b = (int)(row / HW); int hw = (int)(row % HW);
  int h = hw / HPATCH, w = hw % HPATCH;
  int p1 = kk / (PSZ * CC); int rem = kk % (PSZ * CC);
  int p2 = rem / CC; int c = rem % CC;
  Xp[idx] = f2bf(img[(((long)b * CC + c) * IMGSZ + h * PSZ + p1) * IMGSZ + w * PSZ + p2]);
}

__global__ void posadd_kernel(float* __restrict__ tok, const float* __restrict__ hpos,
                              const float* __restrict__ wpos) {
  long idx = (long)blockIdx.x * blockDim.x + threadIdx.x;
  if (idx >= (long)BB * HW * DIM) return;
  int d = (int)(idx % DIM);
  long row = idx / DIM;
  int hw = (int)(row % HW);
  int h = hw / HPATCH, w = hw % HPATCH;
  tok[idx] += hpos[h * DIM + d] + wpos[w * DIM + d];
}

__global__ void slots_init_kernel(const float* __restrict__ mu, const float* __restrict__ logsig,
                                  const float* __restrict__ noise, float* __restrict__ slots) {
  long idx = (long)blockIdx.x * blockDim.x + threadIdx.x;
  if (idx >= (long)BB * NSLOT * DIM) return;
  int d = (int)(idx % DIM);
  slots[idx] = mu[d] + __expf(logsig[d]) * noise[idx];
}

// snapshot f32 + bf16 mirror
__global__ void snapshot_kernel(const float* __restrict__ src, float* __restrict__ dst,
                                __bf16* __restrict__ dst16, long n) {
  long idx = (long)blockIdx.x * blockDim.x + threadIdx.x;
  if (idx < n) { float v = src[idx]; dst[idx] = v; dst16[idx] = f2bf(v); }
}

__global__ __launch_bounds__(256)
void layernorm_kernel(const float* __restrict__ x, const float* __restrict__ g,
                      const float* __restrict__ b, __bf16* __restrict__ y, int dim) {
  __shared__ float red[256];
  long row = blockIdx.x;
  const float* xr = x + row * dim;
  float s = 0.f;
  for (int i = threadIdx.x; i < dim; i += 256) s += xr[i];
  red[threadIdx.x] = s; __syncthreads();
  for (int o = 128; o > 0; o >>= 1) { if ((int)threadIdx.x < o) red[threadIdx.x] += red[threadIdx.x + o]; __syncthreads(); }
  float mean = red[0] / dim; __syncthreads();
  float sv = 0.f;
  for (int i = threadIdx.x; i < dim; i += 256) { float d = xr[i] - mean; sv += d * d; }
  red[threadIdx.x] = sv; __syncthreads();
  for (int o = 128; o > 0; o >>= 1) { if ((int)threadIdx.x < o) red[threadIdx.x] += red[threadIdx.x + o]; __syncthreads(); }
  float rstd = rsqrtf(red[0] / dim + 1e-5f);
  for (int i = threadIdx.x; i < dim; i += 256)
    y[row * dim + i] = f2bf((xr[i] - mean) * rstd * g[i] + b[i]);
}

// lucidrains RMSNorm: x / max(||x||, 1e-12) * sqrt(dim) * g   (bf16 out, GEMM-only consumer)
__global__ __launch_bounds__(256)
void rmsnorm_kernel(const float* __restrict__ x, const float* __restrict__ g,
                    __bf16* __restrict__ y, int dim) {
  __shared__ float red[256];
  long row = blockIdx.x;
  const float* xr = x + row * dim;
  float s = 0.f;
  for (int i = threadIdx.x; i < dim; i += 256) { float v = xr[i]; s += v * v; }
  red[threadIdx.x] = s; __syncthreads();
  for (int o = 128; o > 0; o >>= 1) { if ((int)threadIdx.x < o) red[threadIdx.x] += red[threadIdx.x + o]; __syncthreads(); }
  float scale = sqrtf((float)dim) / fmaxf(sqrtf(red[0]), 1e-12f);
  for (int i = threadIdx.x; i < dim; i += 256)
    y[row * dim + i] = f2bf(xr[i] * scale * g[i]);
}

// row softmax in-place (f32) + bf16 mirror with padded row stride
__global__ __launch_bounds__(256)
void softmax_rows_kernel(float* __restrict__ x, int ncols, __bf16* __restrict__ y16, int ld16) {
  __shared__ float red[256];
  float* xr = x + (long)blockIdx.x * ncols;
  __bf16* yr = y16 + (long)blockIdx.x * ld16;
  float m = -1e30f;
  for (int i = threadIdx.x; i < ncols; i += 256) m = fmaxf(m, xr[i]);
  red[threadIdx.x] = m; __syncthreads();
  for (int o = 128; o > 0; o >>= 1) { if ((int)threadIdx.x < o) red[threadIdx.x] = fmaxf(red[threadIdx.x], red[threadIdx.x + o]); __syncthreads(); }
  m = red[0]; __syncthreads();
  float s = 0.f;
  for (int i = threadIdx.x; i < ncols; i += 256) { float e = __expf(xr[i] - m); xr[i] = e; s += e; }
  red[threadIdx.x] = s; __syncthreads();
  for (int o = 128; o > 0; o >>= 1) { if ((int)threadIdx.x < o) red[threadIdx.x] += red[threadIdx.x + o]; __syncthreads(); }
  float inv = 1.f / red[0];
  for (int i = threadIdx.x; i < ncols; i += 256) { float v = xr[i] * inv; xr[i] = v; yr[i] = f2bf(v); }
  for (int i = ncols + threadIdx.x; i < ld16; i += 256) yr[i] = f2bf(0.f);  // zero pad (K tail)
}

// slot attention: softmax over slots (axis i), +1e-8, normalize over inputs (axis j); bf16 mirror
__global__ __launch_bounds__(256)
void slot_softmax_kernel(float* __restrict__ dots, __bf16* __restrict__ a16) {
  __shared__ float rowinv[NSLOT];
  float* d = dots + (long)blockIdx.x * NSLOT * HW;
  __bf16* a = a16 + (long)blockIdx.x * NSLOT * HW;
  int j = threadIdx.x;          // HW == 256 == blockDim
  float m = -1e30f;
  for (int i2 = 0; i2 < NSLOT; i2++) m = fmaxf(m, d[i2 * HW + j]);
  float s = 0.f;
  for (int i2 = 0; i2 < NSLOT; i2++) { float e = __expf(d[i2 * HW + j] - m); d[i2 * HW + j] = e; s += e; }
  float inv = 1.f / s;
  for (int i2 = 0; i2 < NSLOT; i2++) d[i2 * HW + j] = d[i2 * HW + j] * inv + 1e-8f;
  __syncthreads();
  if (j < NSLOT) {
    float rs = 0.f;
    for (int jj = 0; jj < HW; jj++) rs += d[j * HW + jj];
    rowinv[j] = 1.f / rs;
  }
  __syncthreads();
  for (int i2 = 0; i2 < NSLOT; i2++) {
    float v = d[i2 * HW + j] * rowinv[i2];
    d[i2 * HW + j] = v; a[i2 * HW + j] = f2bf(v);
  }
}

__global__ void gru_kernel(const float* __restrict__ gx, const float* __restrict__ gh,
                           const float* __restrict__ prev, float* __restrict__ slots,
                           __bf16* __restrict__ slots16) {
  long idx = (long)blockIdx.x * blockDim.x + threadIdx.x;
  if (idx >= (long)BB * NSLOT * DIM) return;
  long row = idx / DIM; int d = (int)(idx % DIM);
  const float* gxr = gx + row * 3 * DIM;
  const float* ghr = gh + row * 3 * DIM;
  float r = sigmoidf_(gxr[d] + ghr[d]);
  float z = sigmoidf_(gxr[DIM + d] + ghr[DIM + d]);
  float n = tanhf(gxr[2 * DIM + d] + r * ghr[2 * DIM + d]);
  float v = (1.f - z) * n + z * prev[idx];
  slots[idx] = v; slots16[idx] = f2bf(v);
}

__global__ void s2c_kernel(const float* __restrict__ obj, const float* __restrict__ w2,
                           const float* __restrict__ b2, float* __restrict__ sc) {
  int idx = blockIdx.x * blockDim.x + threadIdx.x;
  if (idx >= BB * NSLOT * 2) return;
  int j = idx & 1; long row = idx >> 1;
  const float* o = obj + row * DIM;
  float s = b2[j];
  for (int k = 0; k < DIM; k++) s += o[k] * w2[k * 2 + j];
  sc[idx] = s;
}

__global__ void coords_kernel(const float* __restrict__ sc, float* __restrict__ coords) {
  int idx = blockIdx.x * blockDim.x + threadIdx.x;
  if (idx >= BB * NTOK * 2) return;
  int c2 = idx & 1; int rn = idx >> 1;
  int b = rn / NTOK, n = rn % NTOK;
  float v;
  if (n < NSLOT) v = sc[((long)b * NSLOT + n) * 2 + c2];
  else { int t = n - NSLOT; v = (c2 == 0) ? (float)(t / HPATCH) : (float)(t % HPATCH); }
  coords[idx] = v;
}

__global__ void concat_kernel(const float* __restrict__ slots, const float* __restrict__ tok,
                              float* __restrict__ x) {
  long idx = (long)blockIdx.x * blockDim.x + threadIdx.x;
  if (idx >= (long)BB * NTOK * DIM) return;
  int d = (int)(idx % DIM);
  long rn = idx / DIM;
  int b = (int)(rn / NTOK), n = (int)(rn % NTOK);
  x[idx] = (n < NSLOT) ? slots[((long)b * NSLOT + n) * DIM + d]
                       : tok[((long)b * HW + (n - NSLOT)) * DIM + d];
}

// ---------------- fused rel-pos MLP bias: 2 -> 128(silu) -> 128(silu, WMMA) -> 8 ----------------
#define RPT_I 4
#define RPT_J 16
__global__ __launch_bounds__(64)
void relpos_kernel(const float* __restrict__ coords,
                   const float* __restrict__ w1, const float* __restrict__ b1,
                   const __bf16* __restrict__ w2t,   // pre-transposed [N][K] bf16
                   const float* __restrict__ b2,
                   const float* __restrict__ w3, const float* __restrict__ b3,
                   float* __restrict__ bias)
{
  __shared__ __bf16 H[RPT_I * RPT_J][RPD];   // 64 x 128 bf16 (h1, then h2 in place)
  __shared__ __bf16 W2s[RPD][RPD];           // 32KB, [N][K] — filled by async DMA
  const int b = blockIdx.z;
  const int tid = threadIdx.x;
  const int il = tid >> 4, jl = tid & 15;
  const int gi = blockIdx.y * RPT_I + il;
  const int gj = blockIdx.x * RPT_J + jl;

  // async-copy W2 (2048 x 16B chunks, 64 lanes x 32 iters)
  const unsigned w2base = lds_addr_of(&W2s[0][0]);
  for (int it2 = 0; it2 < (RPD * RPD / 8) / 64; it2++) {
    int c = tid + 64 * it2;
    async_b128(w2base + c * 16, w2t + (long)c * 8);
  }
  // layer 1 (K=2): VALU
  float r0 = 0.f, r1 = 0.f;
  if (gi < NTOK && gj < NTOK) {
    const float* cb = coords + (long)b * NTOK * 2;
    r0 = cb[gi * 2 + 0] - cb[gj * 2 + 0];
    r1 = cb[gi * 2 + 1] - cb[gj * 2 + 1];
  }
  for (int r = 0; r < RPD; r++) {
    float h = r0 * w1[r] + r1 * w1[RPD + r] + b1[r];
    H[tid][r] = f2bf(h * sigmoidf_(h));
  }
  wait_async0();
  __syncthreads();

  // layer 2 (128x128): WMMA; each wave owns its own 32 rows
  const int wv = tid >> 5, lane = tid & 31, hlf = lane >> 4, l16 = lane & 15;
  for (int ch = 0; ch < 2; ch++) {
    int row0 = wv * 32 + ch * 16;
    const floatx8 zero = {0.f, 0.f, 0.f, 0.f, 0.f, 0.f, 0.f, 0.f};
    floatx8 acc[8] = {zero, zero, zero, zero, zero, zero, zero, zero};
#pragma unroll
    for (int s = 0; s < RPD / 32; s++) {
      bf16x16 af = load_frag(&H[row0][s * 32], RPD, hlf, l16);
#pragma unroll
      for (int ni = 0; ni < 8; ni++) {
        bf16x16 bfr = load_frag(&W2s[ni * 16][s * 32], RPD, hlf, l16);
        acc[ni] = __builtin_amdgcn_wmma_f32_16x16x32_bf16(
            false, af, false, bfr, (short)0, acc[ni], false, false);
      }
    }
#pragma unroll
    for (int ni = 0; ni < 8; ni++) {
      int ncol = ni * 16 + l16;
      float bb = b2[ncol];
#pragma unroll
      for (int v = 0; v < 8; v++) {
        int mr = row0 + v + 8 * hlf;
        float hv = acc[ni][v] + bb;
        H[mr][ncol] = f2bf(hv * sigmoidf_(hv));
      }
    }
  }
  __syncthreads();

  // layer 3 (128 -> 8 heads): VALU
  if (gi < NTOK && gj < NTOK) {
    float o[NHEADS];
#pragma unroll
    for (int hh = 0; hh < NHEADS; hh++) o[hh] = b3[hh];
    for (int k = 0; k < RPD; k++) {
      float hv = bf2f(H[tid][k]);
#pragma unroll
      for (int hh = 0; hh < NHEADS; hh++) o[hh] += hv * w3[k * NHEADS + hh];
    }
#pragma unroll
    for (int hh = 0; hh < NHEADS; hh++)
      bias[(((long)b * NHEADS + hh) * NTOK + gi) * NTOK + gj] = o[hh];
  }
}

// ---------------- host-side launch helpers ----------------
static inline void gemm(hipStream_t s,
                        const __bf16* A, int lda, long sAb, long sAh,
                        const __bf16* B, int ldb, long sBb, long sBh,
                        const float* bias,
                        const float* Res, int ldr, long sRb, long sRh,
                        float* C, __bf16* C16, int ldc, long sCb, long sCh,
                        int M, int N, int K, int batches, int batchH,
                        float alpha, int act, int transB) {
  dim3 grid((N + TNN - 1) / TNN, (M + TMM - 1) / TMM, batches);
  gemm_kernel<<<grid, 128, 0, s>>>(A, lda, sAb, sAh, B, ldb, sBb, sBh, bias,
                                   Res, ldr, sRb, sRh, C, C16, ldc, sCb, sCh,
                                   M, N, K, batchH, alpha, act, transB);
}
static inline int gs(long n) { return (int)((n + 255) / 256); }

extern "C" void kernel_launch(void* const* d_in, const int* in_sizes, int n_in,
                              void* d_out, int out_size, void* d_ws, size_t ws_size,
                              hipStream_t stream) {
  (void)in_sizes; (void)n_in; (void)out_size; (void)ws_size;
  int i = 0;
  const float* images   = (const float*)d_in[i++];
  const float* noise    = (const float*)d_in[i++];
  const float* patch_w  = (const float*)d_in[i++];
  const float* patch_b  = (const float*)d_in[i++];
  const float* hpos     = (const float*)d_in[i++];
  const float* wpos     = (const float*)d_in[i++];
  const float* rp_w1    = (const float*)d_in[i++];
  const float* rp_b1    = (const float*)d_in[i++];
  const float* rp_w2    = (const float*)d_in[i++];
  const float* rp_b2    = (const float*)d_in[i++];
  const float* rp_w3    = (const float*)d_in[i++];
  const float* rp_b3    = (const float*)d_in[i++];
  const float* slots_mu = (const float*)d_in[i++];
  const float* slots_ls = (const float*)d_in[i++];
  const float* ln_in_g  = (const float*)d_in[i++];
  const float* ln_in_b  = (const float*)d_in[i++];
  const float* ln_sl_g  = (const float*)d_in[i++];
  const float* ln_sl_b  = (const float*)d_in[i++];
  const float* sa_wq    = (const float*)d_in[i++];
  const float* sa_bq    = (const float*)d_in[i++];
  const float* sa_wk    = (const float*)d_in[i++];
  const float* sa_bk    = (const float*)d_in[i++];
  const float* sa_wv    = (const float*)d_in[i++];
  const float* sa_bv    = (const float*)d_in[i++];
  const float* sa_wo    = (const float*)d_in[i++];
  const float* sa_bo    = (const float*)d_in[i++];
  const float* gru_wih  = (const float*)d_in[i++];
  const float* gru_whh  = (const float*)d_in[i++];
  const float* gru_bih  = (const float*)d_in[i++];
  const float* gru_bhh  = (const float*)d_in[i++];
  const float* ln_ff_g  = (const float*)d_in[i++];
  const float* ln_ff_b  = (const float*)d_in[i++];
  const float* mlp_w1   = (const float*)d_in[i++];
  const float* mlp_b1   = (const float*)d_in[i++];
  const float* mlp_w2   = (const float*)d_in[i++];
  const float* mlp_b2   = (const float*)d_in[i++];
  const float* s2c_w    = (const float*)d_in[i++];
  const float* s2c_b    = (const float*)d_in[i++];
  const float* attn_g   = (const float*)d_in[i++];
  const float* wq       = (const float*)d_in[i++];
  const float* wk       = (const float*)d_in[i++];
  const float* wv       = (const float*)d_in[i++];
  const float* wo       = (const float*)d_in[i++];
  const float* ff_g     = (const float*)d_in[i++];
  const float* ff_w1    = (const float*)d_in[i++];
  const float* ff_b1    = (const float*)d_in[i++];
  const float* ff_w2    = (const float*)d_in[i++];
  const float* ff_b2    = (const float*)d_in[i++];
  const float* final_g  = (const float*)d_in[i++];
  const float* pred_w   = (const float*)d_in[i++];
  const float* pred_b   = (const float*)d_in[i++];

  // ---- workspace carve-up (64B aligned slices) ----
  char* base = (char*)d_ws;
  size_t off = 0;
  auto allocF = [&](long n) { float* p = (float*)(base + off); off += (((size_t)n * 4) + 63) & ~(size_t)63; return p; };
  auto allocH = [&](long n) { __bf16* p = (__bf16*)(base + off); off += (((size_t)n * 2) + 63) & ~(size_t)63; return p; };

  // f32 masters
  float* tok    = allocF((long)BB * HW * DIM);
  float* slots  = allocF((long)BB * NSLOT * DIM);
  float* prev   = allocF((long)BB * NSLOT * DIM);
  float* dots   = allocF((long)BB * SHEADS * NSLOT * HW);
  float* gx     = allocF((long)BB * NSLOT * 3 * DIM);
  float* gh     = allocF((long)BB * NSLOT * 3 * DIM);
  float* sc     = allocF((long)BB * NSLOT * 2);
  float* coords = allocF((long)BB * NTOK * 2);
  float* biasb  = allocF((long)BB * NHEADS * NTOK * NTOK);
  float* x      = allocF((long)BB * NTOK * DIM);
  float* sim    = allocF((long)BB * NHEADS * NTOK * NTOK);
  // bf16 activations
  __bf16* Xp16    = allocH((long)BB * HW * PATCH_K);
  __bf16* inp16   = allocH((long)BB * HW * DIM);
  __bf16* kbuf16  = allocH((long)BB * HW * SI);
  __bf16* vbuf16  = allocH((long)BB * HW * SI);
  __bf16* sln16   = allocH((long)BB * NSLOT * DIM);
  __bf16* qs16    = allocH((long)BB * NSLOT * SI);
  __bf16* attn16  = allocH((long)BB * SHEADS * NSLOT * HW);
  __bf16* updc16  = allocH((long)BB * NSLOT * SI);
  __bf16* upd16   = allocH((long)BB * NSLOT * DIM);
  __bf16* ffh16   = allocH((long)BB * NSLOT * DIM);
  __bf16* slots16 = allocH((long)BB * NSLOT * DIM);
  __bf16* prev16  = allocH((long)BB * NSLOT * DIM);
  __bf16* xn16    = allocH((long)BB * NTOK * DIM);
  __bf16* qh16    = allocH((long)BB * NTOK * DIM);
  __bf16* kh16    = allocH((long)BB * NTOK * DIM);
  __bf16* vh16    = allocH((long)BB * NTOK * DIM);
  __bf16* sim16   = allocH((long)BB * NHEADS * NTOK * SIMLD);
  __bf16* obuf16  = allocH((long)BB * NTOK * DIM);
  __bf16* fft16   = allocH((long)BB * NTOK * FFD);
  // bf16 weights
  __bf16* patchw16 = allocH((long)PATCH_K * DIM);
  __bf16* swq16 = allocH((long)DIM * SI);
  __bf16* swk16 = allocH((long)DIM * SI);
  __bf16* swv16 = allocH((long)DIM * SI);
  __bf16* swo16 = allocH((long)SI * DIM);
  __bf16* wih16 = allocH((long)DIM * 3 * DIM);
  __bf16* whh16 = allocH((long)DIM * 3 * DIM);
  __bf16* mw1_16 = allocH((long)DIM * DIM);
  __bf16* mw2_16 = allocH((long)DIM * DIM);
  __bf16* wqt16 = allocH((long)DEPTH * DIM * DIM);
  __bf16* wkt16 = allocH((long)DEPTH * DIM * DIM);
  __bf16* wvt16 = allocH((long)DEPTH * DIM * DIM);
  __bf16* wot16 = allocH((long)DEPTH * DIM * DIM);
  __bf16* fw1_16 = allocH((long)DEPTH * DIM * FFD);
  __bf16* fw2_16 = allocH((long)DEPTH * FFD * DIM);
  __bf16* predw16 = allocH((long)DIM * DIM);
  __bf16* w2t16 = allocH((long)RPD * RPD);

  auto conv = [&](const float* s, __bf16* d, long n) {
    convert_kernel<<<gs(n), 256, 0, stream>>>(s, d, n);
  };
  conv(patch_w, patchw16, (long)PATCH_K * DIM);
  conv(sa_wq, swq16, (long)DIM * SI);  conv(sa_wk, swk16, (long)DIM * SI);
  conv(sa_wv, swv16, (long)DIM * SI);  conv(sa_wo, swo16, (long)SI * DIM);
  conv(gru_wih, wih16, (long)DIM * 3 * DIM);  conv(gru_whh, whh16, (long)DIM * 3 * DIM);
  conv(mlp_w1, mw1_16, (long)DIM * DIM);  conv(mlp_w2, mw2_16, (long)DIM * DIM);
  conv(wq, wqt16, (long)DEPTH * DIM * DIM);  conv(wk, wkt16, (long)DEPTH * DIM * DIM);
  conv(wv, wvt16, (long)DEPTH * DIM * DIM);  conv(wo, wot16, (long)DEPTH * DIM * DIM);
  conv(ff_w1, fw1_16, (long)DEPTH * DIM * FFD);  conv(ff_w2, fw2_16, (long)DEPTH * FFD * DIM);
  conv(pred_w, predw16, (long)DIM * DIM);
  rpw2t_kernel<<<gs(RPD * RPD), 256, 0, stream>>>(rp_w2, w2t16);

  const float sscale = 0.125f;  // SDH^-0.5
  const float ascale = 0.125f;  // DH^-0.5

  // ---- patch embed + positions ----
  patchify_kernel<<<gs((long)BB * HW * PATCH_K), 256, 0, stream>>>(images, Xp16);
  gemm(stream, Xp16, PATCH_K, 0, 0, patchw16, DIM, 0, 0, patch_b, nullptr, 0, 0, 0,
       tok, nullptr, DIM, 0, 0, BB * HW, DIM, PATCH_K, 1, 1, 1.f, 0, 0);
  posadd_kernel<<<gs((long)BB * HW * DIM), 256, 0, stream>>>(tok, hpos, wpos);

  // ---- slot attention: fixed k, v ----
  layernorm_kernel<<<BB * HW, 256, 0, stream>>>(tok, ln_in_g, ln_in_b, inp16, DIM);
  gemm(stream, inp16, DIM, 0, 0, swk16, SI, 0, 0, sa_bk, nullptr, 0, 0, 0,
       nullptr, kbuf16, SI, 0, 0, BB * HW, SI, DIM, 1, 1, 1.f, 0, 0);
  gemm(stream, inp16, DIM, 0, 0, swv16, SI, 0, 0, sa_bv, nullptr, 0, 0, 0,
       nullptr, vbuf16, SI, 0, 0, BB * HW, SI, DIM, 1, 1, 1.f, 0, 0);
  slots_init_kernel<<<gs((long)BB * NSLOT * DIM), 256, 0, stream>>>(slots_mu, slots_ls, noise, slots);

  for (int it = 0; it < ITERS; it++) {
    snapshot_kernel<<<gs((long)BB * NSLOT * DIM), 256, 0, stream>>>(slots, prev, prev16, (long)BB * NSLOT * DIM);
    layernorm_kernel<<<BB * NSLOT, 256, 0, stream>>>(slots, ln_sl_g, ln_sl_b, sln16, DIM);
    gemm(stream, sln16, DIM, 0, 0, swq16, SI, 0, 0, sa_bq, nullptr, 0, 0, 0,
         nullptr, qs16, SI, 0, 0, BB * NSLOT, SI, DIM, 1, 1, 1.f, 0, 0);
    // dots[b,h,i,j] = scale * q . k^T
    gemm(stream, qs16, SI, (long)NSLOT * SI, SDH,
         kbuf16, SI, (long)HW * SI, SDH, nullptr, nullptr, 0, 0, 0,
         dots, nullptr, HW, (long)SHEADS * NSLOT * HW, (long)NSLOT * HW,
         NSLOT, HW, SDH, BB * SHEADS, SHEADS, sscale, 0, 1);
    slot_softmax_kernel<<<BB * SHEADS, 256, 0, stream>>>(dots, attn16);
    // updc[b,i,h*64+d] = attn @ v
    gemm(stream, attn16, HW, (long)SHEADS * NSLOT * HW, (long)NSLOT * HW,
         vbuf16, SI, (long)HW * SI, SDH, nullptr, nullptr, 0, 0, 0,
         nullptr, updc16, SI, (long)NSLOT * SI, SDH,
         NSLOT, SDH, HW, BB * SHEADS, SHEADS, 1.f, 0, 0);
    gemm(stream, updc16, SI, 0, 0, swo16, DIM, 0, 0, sa_bo, nullptr, 0, 0, 0,
         nullptr, upd16, DIM, 0, 0, BB * NSLOT, DIM, SI, 1, 1, 1.f, 0, 0);
    // GRU
    gemm(stream, upd16, DIM, 0, 0, wih16, 3 * DIM, 0, 0, gru_bih, nullptr, 0, 0, 0,
         gx, nullptr, 3 * DIM, 0, 0, BB * NSLOT, 3 * DIM, DIM, 1, 1, 1.f, 0, 0);
    gemm(stream, prev16, DIM, 0, 0, whh16, 3 * DIM, 0, 0, gru_bhh, nullptr, 0, 0, 0,
         gh, nullptr, 3 * DIM, 0, 0, BB * NSLOT, 3 * DIM, DIM, 1, 1, 1.f, 0, 0);
    gru_kernel<<<gs((long)BB * NSLOT * DIM), 256, 0, stream>>>(gx, gh, prev, slots, slots16);
    // slot FFN (residual)
    layernorm_kernel<<<BB * NSLOT, 256, 0, stream>>>(slots, ln_ff_g, ln_ff_b, sln16, DIM);
    gemm(stream, sln16, DIM, 0, 0, mw1_16, DIM, 0, 0, mlp_b1, nullptr, 0, 0, 0,
         nullptr, ffh16, DIM, 0, 0, BB * NSLOT, DIM, DIM, 1, 1, 1.f, 1, 0);
    gemm(stream, ffh16, DIM, 0, 0, mw2_16, DIM, 0, 0, mlp_b2, slots, DIM, 0, 0,
         slots, slots16, DIM, 0, 0, BB * NSLOT, DIM, DIM, 1, 1, 1.f, 0, 0);
  }

  // ---- rel-pos bias ----
  s2c_kernel<<<gs(BB * NSLOT * 2), 256, 0, stream>>>(slots, s2c_w, s2c_b, sc);
  coords_kernel<<<gs(BB * NTOK * 2), 256, 0, stream>>>(sc, coords);
  {
    dim3 g((NTOK + RPT_J - 1) / RPT_J, (NTOK + RPT_I - 1) / RPT_I, BB);
    relpos_kernel<<<g, RPT_I * RPT_J, 0, stream>>>(coords, rp_w1, rp_b1, w2t16, rp_b2,
                                                   rp_w3, rp_b3, biasb);
  }

  // ---- transformer ----
  concat_kernel<<<gs((long)BB * NTOK * DIM), 256, 0, stream>>>(slots, tok, x);
  const long sXb = (long)NTOK * DIM;
  const long sBh_nn = (long)NTOK * NTOK;
  const long sBb_nn = (long)NHEADS * NTOK * NTOK;
  const long sSh16 = (long)NTOK * SIMLD;
  const long sSb16 = (long)NHEADS * NTOK * SIMLD;
  for (int l = 0; l < DEPTH; l++) {
    rmsnorm_kernel<<<BB * NTOK, 256, 0, stream>>>(x, attn_g + (long)l * DIM, xn16, DIM);
    const __bf16* wql = wqt16 + (long)l * DIM * DIM;
    const __bf16* wkl = wkt16 + (long)l * DIM * DIM;
    const __bf16* wvl = wvt16 + (long)l * DIM * DIM;
    const __bf16* wol = wot16 + (long)l * DIM * DIM;
    gemm(stream, xn16, DIM, 0, 0, wql, DIM, 0, 0, nullptr, nullptr, 0, 0, 0,
         nullptr, qh16, DIM, 0, 0, BB * NTOK, DIM, DIM, 1, 1, 1.f, 0, 0);
    gemm(stream, xn16, DIM, 0, 0, wkl, DIM, 0, 0, nullptr, nullptr, 0, 0, 0,
         nullptr, kh16, DIM, 0, 0, BB * NTOK, DIM, DIM, 1, 1, 1.f, 0, 0);
    gemm(stream, xn16, DIM, 0, 0, wvl, DIM, 0, 0, nullptr, nullptr, 0, 0, 0,
         nullptr, vh16, DIM, 0, 0, BB * NTOK, DIM, DIM, 1, 1, 1.f, 0, 0);
    // sim = ascale * q.k^T + bias
    gemm(stream, qh16, DIM, sXb, DH, kh16, DIM, sXb, DH, nullptr,
         biasb, NTOK, sBb_nn, sBh_nn,
         sim, nullptr, NTOK, sBb_nn, sBh_nn,
         NTOK, NTOK, DH, BB * NHEADS, NHEADS, ascale, 0, 1);
    softmax_rows_kernel<<<BB * NHEADS * NTOK, 256, 0, stream>>>(sim, NTOK, sim16, SIMLD);
    // o = attn @ v
    gemm(stream, sim16, SIMLD, sSb16, sSh16, vh16, DIM, sXb, DH, nullptr,
         nullptr, 0, 0, 0, nullptr, obuf16, DIM, sXb, DH,
         NTOK, DH, NTOK, BB * NHEADS, NHEADS, 1.f, 0, 0);
    // x = x + o @ wo
    gemm(stream, obuf16, DIM, 0, 0, wol, DIM, 0, 0, nullptr, x, DIM, 0, 0,
         x, nullptr, DIM, 0, 0, BB * NTOK, DIM, DIM, 1, 1, 1.f, 0, 0);
    // FFN
    rmsnorm_kernel<<<BB * NTOK, 256, 0, stream>>>(x, ff_g + (long)l * DIM, xn16, DIM);
    gemm(stream, xn16, DIM, 0, 0, fw1_16 + (long)l * DIM * FFD, FFD, 0, 0,
         ff_b1 + (long)l * FFD, nullptr, 0, 0, 0,
         nullptr, fft16, FFD, 0, 0, BB * NTOK, FFD, DIM, 1, 1, 1.f, 2, 0);
    gemm(stream, fft16, FFD, 0, 0, fw2_16 + (long)l * FFD * DIM, DIM, 0, 0,
         ff_b2 + (long)l * DIM, x, DIM, 0, 0,
         x, nullptr, DIM, 0, 0, BB * NTOK, DIM, FFD, 1, 1, 1.f, 0, 0);
  }

  // ---- final norm + prediction head on tokens only ----
  rmsnorm_kernel<<<BB * NTOK, 256, 0, stream>>>(x, final_g, xn16, DIM);
  gemm(stream, xn16 + (long)NSLOT * DIM, DIM, sXb, 0,
       predw16, DIM, 0, 0, pred_b, nullptr, 0, 0, 0,
       (float*)d_out, nullptr, DIM, (long)HW * DIM, 0,
       HW, DIM, DIM, BB, 1, 1.f, 0, 0);
}
